// AutoregressiveContextEncoder_79714593014160
// MI455X (gfx1250) — compile-verified, matrix operand order checked by reading, not run on hardware
//
#include <hip/hip_runtime.h>
#include <hip/hip_bf16.h>

typedef __attribute__((ext_vector_type(16))) __bf16 v16bf;
typedef __attribute__((ext_vector_type(8)))  __bf16 v8bf;
typedef __attribute__((ext_vector_type(8)))  float  v8f;

// ---------------------------------------------------------------------------
// Async global->LDS copy (CDNA5 GLOBAL_LOAD_ASYNC_TO_LDS_*, ASYNCcnt-tracked)
// Builtin signature (from hipcc diagnostics):
//   b128: (AS1 int4* gsrc, AS3 int4* ldst, imm offset, imm cpol)
//   b32 : (AS1 int*  gsrc, AS3 int*  ldst, imm offset, imm cpol)
// ---------------------------------------------------------------------------
#if __has_builtin(__builtin_amdgcn_global_load_async_to_lds_b128) && \
    __has_builtin(__builtin_amdgcn_global_load_async_to_lds_b32) &&  \
    __has_builtin(__builtin_amdgcn_s_wait_asynccnt)
#define USE_ASYNC_LDS 1
#else
#define USE_ASYNC_LDS 0
#endif

#if USE_ASYNC_LDS
typedef int v4i_ __attribute__((vector_size(16)));
typedef __attribute__((address_space(1))) v4i_* g128p;
typedef __attribute__((address_space(3))) v4i_* l128p;
typedef __attribute__((address_space(1))) int*  g32p;
typedef __attribute__((address_space(3))) int*  l32p;
__device__ __forceinline__ void cp_g2l_b128(void* l, const void* g) {
    __builtin_amdgcn_global_load_async_to_lds_b128((g128p)g, (l128p)l, 0, 0);
}
__device__ __forceinline__ void cp_g2l_b32(void* l, const void* g) {
    __builtin_amdgcn_global_load_async_to_lds_b32((g32p)g, (l32p)l, 0, 0);
}
#define CP_WAIT(n) __builtin_amdgcn_s_wait_asynccnt(n)
#else
__device__ __forceinline__ void cp_g2l_b128(void* l, const void* g) {
    *(v8bf*)l = *(const v8bf*)g;
}
__device__ __forceinline__ void cp_g2l_b32(void* l, const void* g) {
    *(unsigned int*)l = *(const unsigned int*)g;
}
#define CP_WAIT(n) ((void)0)
#endif

// ---------------------------------------------------------------------------
// Conversion helpers
// ---------------------------------------------------------------------------
__global__ void f32_to_bf16_kernel(const float* __restrict__ src,
                                   __bf16* __restrict__ dst, size_t n) {
    size_t i = (size_t)blockIdx.x * blockDim.x + threadIdx.x;
    if (i < n) dst[i] = (__bf16)src[i];
}

__global__ void pad_convert_kernel(const float* __restrict__ src,
                                   __bf16* __restrict__ dst,
                                   int rows, int src_cols, int dst_cols) {
    size_t i = (size_t)blockIdx.x * blockDim.x + threadIdx.x;
    size_t total = (size_t)rows * dst_cols;
    if (i >= total) return;
    int r = (int)(i / dst_cols);
    int c = (int)(i % dst_cols);
    dst[i] = (c < src_cols) ? (__bf16)src[(size_t)r * src_cols + c] : (__bf16)0.0f;
}

__global__ void zero_h_kernel(float* __restrict__ hf, __bf16* __restrict__ hb, int n) {
    int i = blockIdx.x * blockDim.x + threadIdx.x;
    if (i < n) { hf[i] = 0.0f; hb[i] = (__bf16)0.0f; }
}

// ---------------------------------------------------------------------------
// bf16 WMMA GEMM, fp32 accumulate:  C[M x N] = A[M x K] * W[N x K]^T + bias[N]
// Block: 256 threads (8 waves). Block tile 64x128, K step 32, double-buffered
// LDS filled with async global->LDS copies. Each wave owns a 32x32 output
// (2 a-frags x 2 b-frags -> 4 v_wmma per K chunk).
// ---------------------------------------------------------------------------
__global__ __launch_bounds__(256)
void gemm_bias_kernel(const __bf16* __restrict__ A,
                      const __bf16* __restrict__ W,
                      const float*  __restrict__ bias,
                      float*   __restrict__ Cf,    // optional fp32 out (ld = N)
                      __bf16*  __restrict__ Cb,    // optional bf16 out (ld = N)
                      int M, int N, int K) {
    __shared__ alignas(16) __bf16 As[2][64][32];
    __shared__ alignas(16) __bf16 Ws[2][128][32];

    const int m0   = blockIdx.y * 64;
    const int n0   = blockIdx.x * 128;
    const int tid  = threadIdx.x;
    const int wave = tid >> 5;
    const int lane = tid & 31;
    const int wm   = wave >> 2;   // 0..1 -> 32-row sub tile
    const int wn   = wave & 3;    // 0..3 -> 32-col sub tile
    const int half = lane >> 4;
    const int l15  = lane & 15;
    const int kb   = half * 8;    // ISA 16-bit A/B fragment K layout

    // per-thread copy assignments
    const int ar = tid >> 2;            // A: row 0..63
    const int ac = (tid & 3) * 8;       //    col chunk (8 bf16 = 16B)
    const int w0r = (tid * 2) >> 2;     // W: two 16B chunks per thread
    const int w0c = ((tid * 2) & 3) * 8;
    const int w1r = (tid * 2 + 1) >> 2;
    const int w1c = ((tid * 2 + 1) & 3) * 8;

    const int nchunk = K >> 5;

    // prefetch chunk 0 into buffer 0
    cp_g2l_b128(&As[0][ar][ac],  A + (size_t)(m0 + ar)  * K + ac);
    cp_g2l_b128(&Ws[0][w0r][w0c], W + (size_t)(n0 + w0r) * K + w0c);
    cp_g2l_b128(&Ws[0][w1r][w1c], W + (size_t)(n0 + w1r) * K + w1c);

    v8f acc[2][2] = {};
    for (int i = 0; i < nchunk; ++i) {
        const int p = i & 1;
        if (i + 1 < nchunk) {
            const int k1 = (i + 1) << 5;
            cp_g2l_b128(&As[1 - p][ar][ac],   A + (size_t)(m0 + ar)  * K + k1 + ac);
            cp_g2l_b128(&Ws[1 - p][w0r][w0c], W + (size_t)(n0 + w0r) * K + k1 + w0c);
            cp_g2l_b128(&Ws[1 - p][w1r][w1c], W + (size_t)(n0 + w1r) * K + k1 + w1c);
            CP_WAIT(3);                 // chunk i complete (in-order)
        } else {
            CP_WAIT(0);
        }
        __syncthreads();                // all waves' chunk-i data visible

        v16bf a[2], b[2];
#pragma unroll
        for (int mi = 0; mi < 2; ++mi) {
            const int r = wm * 32 + mi * 16 + l15;
#pragma unroll
            for (int j = 0; j < 8; ++j) {
                a[mi][j]     = As[p][r][kb + j];
                a[mi][8 + j] = As[p][r][kb + 16 + j];
            }
        }
#pragma unroll
        for (int ni = 0; ni < 2; ++ni) {
            const int r = wn * 32 + ni * 16 + l15;
#pragma unroll
            for (int j = 0; j < 8; ++j) {
                b[ni][j]     = Ws[p][r][kb + j];
                b[ni][8 + j] = Ws[p][r][kb + 16 + j];
            }
        }
#pragma unroll
        for (int mi = 0; mi < 2; ++mi)
#pragma unroll
            for (int ni = 0; ni < 2; ++ni)
                acc[mi][ni] = __builtin_amdgcn_wmma_f32_16x16x32_bf16(
                    false, a[mi], false, b[ni], (short)0, acc[mi][ni], false, false);
        __syncthreads();                // safe to overwrite buffer p
    }

#pragma unroll
    for (int ni = 0; ni < 2; ++ni) {
        const int n = n0 + wn * 32 + ni * 16 + l15;
        const float bv = bias ? bias[n] : 0.0f;
#pragma unroll
        for (int mi = 0; mi < 2; ++mi) {
#pragma unroll
            for (int i = 0; i < 8; ++i) {
                const int m = m0 + wm * 32 + mi * 16 + i + half * 8;
                const float v = acc[mi][ni][i] + bv;
                const size_t off = (size_t)m * N + n;
                if (Cf) Cf[off] = v;
                if (Cb) Cb[off] = (__bf16)v;
            }
        }
    }
}

// ---------------------------------------------------------------------------
// Fused GRU recurrent step (one timestep), double-buffered async h staging.
// Wave = 16(batch) x 16(hcol) tile with 3 gate accumulators (3 wmma / chunk).
// Grid: (H/128, B/16) = (8, 8), 256 threads.
// ---------------------------------------------------------------------------
__device__ __forceinline__ v16bf load_w_frag(const __bf16* __restrict__ row, int kb) {
    v8bf lo = *(const v8bf*)(row + kb);
    v8bf hi = *(const v8bf*)(row + kb + 16);
    v16bf b;
#pragma unroll
    for (int j = 0; j < 8; ++j) { b[j] = lo[j]; b[8 + j] = hi[j]; }
    return b;
}

__global__ __launch_bounds__(256)
void gru_step_kernel(const __bf16* __restrict__ h_bf_in,   // B x H
                     const float*  __restrict__ h_f_in,    // B x H
                     __bf16* __restrict__ h_bf_out,
                     float*  __restrict__ h_f_out,
                     const __bf16* __restrict__ Whh,       // 3H x H (this layer)
                     const float*  __restrict__ bhh,       // 3H
                     const float*  __restrict__ xg,        // (B*T) x 3H (with b_ih)
                     const unsigned char* __restrict__ mask, // B x T
                     __bf16* __restrict__ y,               // (B*T) x H
                     int t) {
    const int H = 1024, T = 196;
    __shared__ alignas(16) __bf16 As[2][16][32];

    const int m0   = blockIdx.y * 16;          // batch tile
    const int n0   = blockIdx.x * 128;         // block's h-col tile
    const int tid  = threadIdx.x;
    const int wave = tid >> 5;
    const int lane = tid & 31;
    const int half = lane >> 4;
    const int l15  = lane & 15;
    const int kb   = half * 8;
    const int n    = n0 + wave * 16 + l15;     // this lane's h column

    const __bf16* w0 = Whh + (size_t)(0 * H + n) * H;
    const __bf16* w1 = Whh + (size_t)(1 * H + n) * H;
    const __bf16* w2 = Whh + (size_t)(2 * H + n) * H;

    const int cr = tid >> 4;            // copy row 0..15
    const int cc = (tid & 15) * 2;      // copy col (2 bf16 = 4B)
    const __bf16* hrow = h_bf_in + (size_t)(m0 + cr) * H + cc;

    cp_g2l_b32(&As[0][cr][cc], hrow);   // chunk 0 -> buf 0

    v8f ar = {}, az = {}, an = {};
    for (int i = 0; i < 32; ++i) {      // H/32 chunks
        const int p  = i & 1;
        const int k0 = i << 5;
        if (i + 1 < 32) {
            cp_g2l_b32(&As[1 - p][cr][cc], hrow + (i + 1) * 32);
            CP_WAIT(1);
        } else {
            CP_WAIT(0);
        }
        __syncthreads();

        v16bf a;
#pragma unroll
        for (int j = 0; j < 8; ++j) {
            a[j]     = As[p][l15][kb + j];
            a[8 + j] = As[p][l15][kb + 16 + j];
        }
        __builtin_prefetch(w0 + k0 + 32, 0, 3);
        v16bf br = load_w_frag(w0 + k0, kb);
        v16bf bz = load_w_frag(w1 + k0, kb);
        v16bf bn = load_w_frag(w2 + k0, kb);

        ar = __builtin_amdgcn_wmma_f32_16x16x32_bf16(false, a, false, br, (short)0, ar, false, false);
        az = __builtin_amdgcn_wmma_f32_16x16x32_bf16(false, a, false, bz, (short)0, az, false, false);
        an = __builtin_amdgcn_wmma_f32_16x16x32_bf16(false, a, false, bn, (short)0, an, false, false);
        __syncthreads();
    }

    const float br_ = bhh[0 * H + n];
    const float bz_ = bhh[1 * H + n];
    const float bn_ = bhh[2 * H + n];
#pragma unroll
    for (int i = 0; i < 8; ++i) {
        const int b = m0 + i + half * 8;
        const size_t bt = (size_t)b * T + t;
        const float* xrow = xg + bt * (size_t)(3 * H);
        const float xr = xrow[0 * H + n];
        const float xz = xrow[1 * H + n];
        const float xn = xrow[2 * H + n];
        const float hr = ar[i] + br_;
        const float hz = az[i] + bz_;
        const float hn = an[i] + bn_;
        const float r  = 1.0f / (1.0f + __expf(-(xr + hr)));
        const float z  = 1.0f / (1.0f + __expf(-(xz + hz)));
        const float ng = tanhf(xn + r * hn);
        const float hp = h_f_in[(size_t)b * H + n];
        const float hnew = (1.0f - z) * ng + z * hp;
        const bool m = mask[bt] != 0;
        const float hnext = m ? hnew : hp;
        h_f_out[(size_t)b * H + n]  = hnext;
        h_bf_out[(size_t)b * H + n] = (__bf16)hnext;
        y[bt * (size_t)H + n] = m ? (__bf16)hnew : (__bf16)0.0f;
    }
}

// ---------------------------------------------------------------------------
// Host orchestration
// ---------------------------------------------------------------------------
extern "C" void kernel_launch(void* const* d_in, const int* in_sizes, int n_in,
                              void* d_out, int out_size, void* d_ws, size_t ws_size,
                              hipStream_t stream) {
    (void)in_sizes; (void)n_in; (void)out_size; (void)ws_size;
    const int B = 128, T = 196, DIN = 263, DINP = 288, H = 1024, L = 3;
    const size_t BT = (size_t)B * T;            // 25088 = 64 * 392

    const float* motion = (const float*)d_in[0];
    const unsigned char* mask = (const unsigned char*)d_in[1];   // bool (B,T)
    const float* Wp   = (const float*)d_in[2];
    const float* bp   = (const float*)d_in[3];
    const float* W_ih = (const float*)d_in[4];
    const float* W_hh = (const float*)d_in[5];
    const float* b_ih = (const float*)d_in[6];
    const float* b_hh = (const float*)d_in[7];
    const float* Wo   = (const float*)d_in[8];
    const float* bo   = (const float*)d_in[9];
    float* out = (float*)d_out;

    char* ws = (char*)d_ws;
    size_t off = 0;
    auto alloc = [&](size_t bytes) -> void* {
        void* p = ws + off;
        off = (off + bytes + 255) & ~(size_t)255;
        return p;
    };
    __bf16* motion_b = (__bf16*)alloc(BT * DINP * 2);
    __bf16* Wp_b     = (__bf16*)alloc((size_t)H * DINP * 2);
    __bf16* Wih_b    = (__bf16*)alloc((size_t)L * 3 * H * H * 2);
    __bf16* Whh_b    = (__bf16*)alloc((size_t)L * 3 * H * H * 2);
    __bf16* Wo_b     = (__bf16*)alloc((size_t)H * H * 2);
    float*  xgbuf    = (float*) alloc(BT * 3 * H * 4);
    __bf16* xb0      = (__bf16*)alloc(BT * H * 2);
    __bf16* xb1      = (__bf16*)alloc(BT * H * 2);
    float*  hf       = (float*) alloc((size_t)2 * B * H * 4);
    __bf16* hb       = (__bf16*)alloc((size_t)2 * B * H * 2);

    {
        size_t n = BT * DINP;
        pad_convert_kernel<<<(unsigned)((n + 255) / 256), 256, 0, stream>>>(
            motion, motion_b, (int)BT, DIN, DINP);
        n = (size_t)H * DINP;
        pad_convert_kernel<<<(unsigned)((n + 255) / 256), 256, 0, stream>>>(
            Wp, Wp_b, H, DIN, DINP);
        n = (size_t)L * 3 * H * H;
        f32_to_bf16_kernel<<<(unsigned)((n + 255) / 256), 256, 0, stream>>>(W_ih, Wih_b, n);
        f32_to_bf16_kernel<<<(unsigned)((n + 255) / 256), 256, 0, stream>>>(W_hh, Whh_b, n);
        n = (size_t)H * H;
        f32_to_bf16_kernel<<<(unsigned)((n + 255) / 256), 256, 0, stream>>>(Wo, Wo_b, n);
    }

    // input projection: x0 = motion @ Wp^T + bp (bf16 out)
    gemm_bias_kernel<<<dim3(H / 128, (unsigned)(BT / 64)), 256, 0, stream>>>(
        motion_b, Wp_b, bp, nullptr, xb0, (int)BT, H, DINP);

    __bf16* cur = xb0;
    __bf16* nxt = xb1;
    for (int l = 0; l < L; ++l) {
        // xg = x @ W_ih[l]^T + b_ih[l]  (fp32, all timesteps in parallel)
        gemm_bias_kernel<<<dim3(3 * H / 128, (unsigned)(BT / 64)), 256, 0, stream>>>(
            cur, Wih_b + (size_t)l * 3 * H * H, b_ih + (size_t)l * 3 * H,
            xgbuf, nullptr, (int)BT, 3 * H, H);

        zero_h_kernel<<<(B * H) / 256, 256, 0, stream>>>(hf, hb, B * H);

        for (int t = 0; t < T; ++t) {
            int p = t & 1;
            gru_step_kernel<<<dim3(H / 128, B / 16), 256, 0, stream>>>(
                hb + (size_t)p * B * H,       hf + (size_t)p * B * H,
                hb + (size_t)(1 - p) * B * H, hf + (size_t)(1 - p) * B * H,
                Whh_b + (size_t)l * 3 * H * H, b_hh + (size_t)l * 3 * H,
                xgbuf, mask, nxt, t);
        }
        __bf16* tmp = cur; cur = nxt; nxt = tmp;
    }

    // output projection: out = y @ Wo^T + bo (fp32)
    gemm_bias_kernel<<<dim3(H / 128, (unsigned)(BT / 64)), 256, 0, stream>>>(
        cur, Wo_b, bo, out, nullptr, (int)BT, H, H);
}
